// AttentionLayer_79431125172260
// MI455X (gfx1250) — matrix operand order repaired and checked by reading, not compiled, not run on hardware
//
#include <hip/hip_runtime.h>
#include <hip/hip_bf16.h>
#include <stdint.h>

// ---------------------------------------------------------------------------
// MHA layer for MI455X (gfx1250): bf16 WMMA everywhere, f32 accumulate.
// GEMM tiles staged by the Tensor Data Mover (TENSORcnt, double-buffered,
// hardware LDS padding); attention K tiles staged with async-to-LDS loads
// (ASYNCcnt).  x:[2,2048,1024] f32; Wq/Wk/Wv/Wo:[1024,1024]; out f32.
// ---------------------------------------------------------------------------

#define D_MODEL 1024
#define SEQ     2048
#define BATCH   2
#define MTOT    (BATCH * SEQ)   // 4096 rows total
#define NHEAD   16
#define HDIM    64

typedef __attribute__((ext_vector_type(16))) __bf16 v16bf;
typedef __attribute__((ext_vector_type(8)))  float  v8f;
typedef __attribute__((ext_vector_type(4)))  unsigned int u32x4;
typedef __attribute__((ext_vector_type(8)))  int    i32x8;
typedef __attribute__((ext_vector_type(4)))  int    i32x4;

union FragBF {            // 16 bf16 = 8 VGPRs; filled via two 16B chunks
  v16bf v;
  uint4 q[2];
};

static __device__ __forceinline__ unsigned short f2bf(float f) {
  unsigned int u = __float_as_uint(f);
  u += 0x7FFFu + ((u >> 16) & 1u);
  return (unsigned short)(u >> 16);
}

// Low 32 bits of a flat pointer to __shared__ == LDS byte offset.
static __device__ __forceinline__ unsigned lds_off32(const void* p) {
  return (unsigned)(unsigned long long)p;
}

// --- CDNA5 async global->LDS copy (ASYNCcnt), GV addressing mode ----------
static __device__ __forceinline__ void async_ldg_b128(const void* g, void* l) {
  asm volatile("global_load_async_to_lds_b128 %0, %1, off"
               :: "v"(lds_off32(l)), "v"((unsigned long long)g)
               : "memory");
}
static __device__ __forceinline__ void wait_async0() {
  asm volatile("s_wait_asynccnt 0x0" ::: "memory");
}

// --- CDNA5 Tensor Data Mover: 2D tile (tile_dim0=32 bf16 x tile_rows) -----
// D# per CDNA5 ISA: group0 = {count|flags, lds_addr, global_addr,
// type=2}; group1 = sizes/strides.  data_size=1 (2B elems).  LDS padding:
// pad_interval=3 (16 DWORDs = one 32-elem row) + pad_amount=3 (4 DWORDs =
// 8 elems) -> LDS row stride 40 bf16, matching the [.][40] shared arrays.
static __device__ __forceinline__ void tdm_load_tile_2d(
    const unsigned short* gbase, void* lds, int tile_rows, unsigned tensor_rows) {
  unsigned long long ga = (unsigned long long)gbase;
  u32x4 g0;
  g0[0] = 1u;                                   // count=1 (valid), user mode
  g0[1] = lds_off32(lds);                       // lds_addr
  g0[2] = (unsigned)ga;                         // global_addr[31:0]
  g0[3] = (unsigned)(ga >> 32) | (2u << 30);    // global_addr[56:32] | type=2
  i32x8 g1;
  g1[0] = (1 << 16) | (1 << 20) | (3 << 22) | (3 << 25); // data_size=2B, pad
  g1[1] = (int)(((unsigned)D_MODEL & 0xFFFFu) << 16);    // tensor_dim0 lo16
  g1[2] = (int)((((unsigned)D_MODEL >> 16) & 0xFFFFu) |
                ((tensor_rows & 0xFFFFu) << 16));        // dim0 hi | dim1 lo
  g1[3] = (int)(((tensor_rows >> 16) & 0xFFFFu) |
                (32u << 16));                            // dim1 hi | tile_dim0
  g1[4] = tile_rows;                                     // tile_dim1 (dim2=0)
  g1[5] = D_MODEL;                                       // dim0_stride lo32
  g1[6] = 0;                                             // stride hi | d1s lo
  g1[7] = 0;
  i32x4 z4 = {0, 0, 0, 0};
  i32x8 z8 = {0, 0, 0, 0, 0, 0, 0, 0};
  // 6-arg form (clang-23 / therock-10.0 headers): groups 2/3 + spare unused
  __builtin_amdgcn_tensor_load_to_lds(g0, g1, z4, z4, z8, 0);
}

// ---------------------------------------------------------------------------
// fp32 -> bf16 bulk convert (4 elems / thread)
// ---------------------------------------------------------------------------
__global__ void cvt4(const float4* __restrict__ in, ushort4* __restrict__ out, int n4) {
  int i = blockIdx.x * blockDim.x + threadIdx.x;
  if (i < n4) {
    float4 f = in[i];
    ushort4 r;
    r.x = f2bf(f.x); r.y = f2bf(f.y); r.z = f2bf(f.z); r.w = f2bf(f.w);
    out[i] = r;
  }
}

// ---------------------------------------------------------------------------
// y = (x @ W^T + bias) * scale.  Tiles fed by TDM, double-buffered so the
// DMA of tile k+1 overlaps the 4x WMMA chain on tile k.  Block: 256 thr =
// 8 waves, tile 128(M) x 64(N), k-step 32; wave = 2x2 wmma accumulators.
// ---------------------------------------------------------------------------
template <bool OUT_BF16>
__global__ __launch_bounds__(256) void gemm_xwT(
    const unsigned short* __restrict__ A,
    const unsigned short* __restrict__ W,
    const float* __restrict__ bias,
    void* __restrict__ outp,
    float scale) {
  constexpr int K = D_MODEL;
  constexpr int N = D_MODEL;

  __shared__ unsigned short As[2][128][40];  // 32 k + 8 pad (TDM pad feature)
  __shared__ unsigned short Bs[2][64][40];

  const int t    = threadIdx.x;
  const int m0   = blockIdx.y * 128;
  const int n0   = blockIdx.x * 64;
  const int wave = t >> 5;
  const int lane = t & 31;
  const int half = lane >> 4;
  const int l16  = lane & 15;
  const int wm   = (wave >> 1) * 32;
  const int wn   = (wave & 1) * 32;
  const bool issuer = (wave == 0);  // one wave per WG drives the TDM

  v8f acc[2][2];
#pragma unroll
  for (int i = 0; i < 2; i++)
#pragma unroll
    for (int j = 0; j < 2; j++)
#pragma unroll
      for (int e = 0; e < 8; e++) acc[i][j][e] = 0.0f;

  if (issuer) {
    tdm_load_tile_2d(A + (size_t)m0 * K, &As[0][0][0], 128, MTOT);
    tdm_load_tile_2d(W + (size_t)n0 * K, &Bs[0][0][0], 64, D_MODEL);
  }
  __builtin_amdgcn_s_wait_tensorcnt(0);
  __syncthreads();

  int cur = 0;
  for (int k0 = 0; k0 < K; k0 += 32) {
    // prefetch next k-tile via TDM while this one is consumed
    if (k0 + 32 < K && issuer) {
      tdm_load_tile_2d(A + (size_t)m0 * K + k0 + 32, &As[cur ^ 1][0][0], 128, MTOT);
      tdm_load_tile_2d(W + (size_t)n0 * K + k0 + 32, &Bs[cur ^ 1][0][0], 64, D_MODEL);
    }

    // ---- fragments (gfx1250 16-bit A 16x32 / B 32x16 layouts) ----
    FragBF a[2], b[2];
#pragma unroll
    for (int mi = 0; mi < 2; mi++) {
      int rowL = wm + mi * 16 + l16;
      int ksel = half * 8;  // half0: K 0..7 & 16..23 ; half1: K 8..15 & 24..31
      a[mi].q[0] = *(const uint4*)&As[cur][rowL][ksel];
      a[mi].q[1] = *(const uint4*)&As[cur][rowL][16 + ksel];
    }
#pragma unroll
    for (int ni = 0; ni < 2; ni++) {
      int colL = wn + ni * 16 + l16;
      int koff = half * 8;   // half0: K 0..7 & 16..23 ; half1: K 8..15 & 24..31
      b[ni].q[0] = *(const uint4*)&Bs[cur][colL][koff];
      b[ni].q[1] = *(const uint4*)&Bs[cur][colL][koff + 16];
    }
#pragma unroll
    for (int mi = 0; mi < 2; mi++)
#pragma unroll
      for (int ni = 0; ni < 2; ni++)
        acc[mi][ni] = __builtin_amdgcn_wmma_f32_16x16x32_bf16(
            false, a[mi].v, false, b[ni].v, (short)0, acc[mi][ni], false, false);

    __builtin_amdgcn_s_wait_tensorcnt(0);  // next tile landed (no-op last iter)
    __syncthreads();
    cur ^= 1;
  }

  // ---- epilogue: C layout = VGPR r -> row r + 8*half, col = lane&15 ----
#pragma unroll
  for (int mi = 0; mi < 2; mi++) {
#pragma unroll
    for (int ni = 0; ni < 2; ni++) {
      int col = n0 + wn + ni * 16 + l16;
      float bv = bias[col];
#pragma unroll
      for (int r = 0; r < 8; r++) {
        int row = m0 + wm + mi * 16 + r + 8 * half;
        float v = (acc[mi][ni][r] + bv) * scale;
        if (OUT_BF16)
          ((unsigned short*)outp)[(size_t)row * N + col] = f2bf(v);
        else
          ((float*)outp)[(size_t)row * N + col] = v;
      }
    }
  }
}

// ---------------------------------------------------------------------------
// Flash attention per (batch, head).  Grid: (SEQ/128, B*H).  Block = 8 waves;
// each wave owns 16 query rows.  Key blocks of 64; K staged via async-to-LDS
// loads (ASYNCcnt), V staged transposed so every WMMA B-fragment is a
// contiguous b128 LDS load.  Online softmax in registers via wave32
// half-butterfly shuffles.  Q pre-scaled by 1/sqrt(64) at projection time.
// ---------------------------------------------------------------------------
__global__ __launch_bounds__(256) void attn_kernel(
    const unsigned short* __restrict__ Q,
    const unsigned short* __restrict__ Km,
    const unsigned short* __restrict__ Vm,
    unsigned short* __restrict__ ctx) {
  __shared__ unsigned short Ks[64][72];      // [key][dim]
  __shared__ unsigned short Vt[64][72];      // [dim][key]  (transposed)
  __shared__ unsigned short Pb[8][16][72];   // per-wave P tile (bf16)

  const int t    = threadIdx.x;
  const int wave = t >> 5;
  const int lane = t & 31;
  const int half = lane >> 4;
  const int l16  = lane & 15;

  const int bh     = blockIdx.y;
  const int b      = bh >> 4;
  const int h      = bh & 15;
  const int colh   = h * HDIM;
  const int qrow0  = b * SEQ + blockIdx.x * 128;
  const int kvrow0 = b * SEQ;

  // Q fragments for this wave's 16 rows (2 k-steps over the 64-dim head)
  FragBF aq[2];
  {
    int rowL = qrow0 + wave * 16 + l16;
    int ksel = half * 8;
#pragma unroll
    for (int ks = 0; ks < 2; ks++) {
      const unsigned short* p = &Q[(size_t)rowL * D_MODEL + colh + ks * 32];
      aq[ks].q[0] = *(const uint4*)(p + ksel);
      aq[ks].q[1] = *(const uint4*)(p + 16 + ksel);
    }
  }

  v8f o[4];
#pragma unroll
  for (int ni = 0; ni < 4; ni++)
#pragma unroll
    for (int e = 0; e < 8; e++) o[ni][e] = 0.0f;
  float mrun[8], lrun[8];
#pragma unroll
  for (int r = 0; r < 8; r++) { mrun[r] = -3.0e38f; lrun[r] = 0.0f; }

  for (int j = 0; j < SEQ / 64; j++) {
    // ---- stage K async (global->LDS direct) and V transposed (manual) ----
#pragma unroll
    for (int i = 0; i < 2; i++) {
      int c = t + 256 * i;
      int row  = c >> 3;        // key 0..63
      int col8 = (c & 7) * 8;   // dim chunk
      size_t gr = (size_t)(kvrow0 + j * 64 + row) * D_MODEL + colh + col8;
      async_ldg_b128(&Km[gr], &Ks[row][col8]);
      uint4 vv = *(const uint4*)&Vm[gr];
      const unsigned short* vs = (const unsigned short*)&vv;
#pragma unroll
      for (int e = 0; e < 8; e++) Vt[col8 + e][row] = vs[e];
    }
    wait_async0();
    __syncthreads();

    // ---- S = Qs * K^T  (16 x 64 per wave, 8 WMMAs) ----
    v8f s[4];
#pragma unroll
    for (int ni = 0; ni < 4; ni++)
#pragma unroll
      for (int e = 0; e < 8; e++) s[ni][e] = 0.0f;
#pragma unroll
    for (int ks = 0; ks < 2; ks++) {
#pragma unroll
      for (int ni = 0; ni < 4; ni++) {
        FragBF fb;
        int key  = ni * 16 + l16;
        int koff = ks * 32 + half * 8;
        fb.q[0] = *(const uint4*)&Ks[key][koff];
        fb.q[1] = *(const uint4*)&Ks[key][koff + 16];
        s[ni] = __builtin_amdgcn_wmma_f32_16x16x32_bf16(
            false, aq[ks].v, false, fb.v, (short)0, s[ni], false, false);
      }
    }

    // ---- online softmax in registers (row r lives in a 16-lane half) ----
    float alpha[8];
#pragma unroll
    for (int r = 0; r < 8; r++) {
      float m = s[0][r];
#pragma unroll
      for (int ni = 1; ni < 4; ni++) m = fmaxf(m, s[ni][r]);
#pragma unroll
      for (int mk = 1; mk < 16; mk <<= 1) m = fmaxf(m, __shfl_xor(m, mk, 32));
      float nm = fmaxf(mrun[r], m);
      alpha[r] = __expf(mrun[r] - nm);
      float ls = 0.0f;
#pragma unroll
      for (int ni = 0; ni < 4; ni++) {
        float p = __expf(s[ni][r] - nm);
        s[ni][r] = p;
        ls += p;
      }
#pragma unroll
      for (int mk = 1; mk < 16; mk <<= 1) ls += __shfl_xor(ls, mk, 32);
      lrun[r] = lrun[r] * alpha[r] + ls;
      mrun[r] = nm;
    }

    // ---- spill P as bf16 in A-fragment layout (same-wave LDS is in-order,
    //      so no barrier needed before reload) ----
#pragma unroll
    for (int ni = 0; ni < 4; ni++)
#pragma unroll
      for (int r = 0; r < 8; r++)
        Pb[wave][r + 8 * half][ni * 16 + l16] = f2bf(s[ni][r]);

    // ---- rescale running O, then O += P * V (8 WMMAs) ----
#pragma unroll
    for (int ni = 0; ni < 4; ni++)
#pragma unroll
      for (int r = 0; r < 8; r++) o[ni][r] *= alpha[r];

#pragma unroll
    for (int ks = 0; ks < 2; ks++) {
      FragBF fp;
      int ksel = half * 8;
      fp.q[0] = *(const uint4*)&Pb[wave][l16][ks * 32 + ksel];
      fp.q[1] = *(const uint4*)&Pb[wave][l16][ks * 32 + 16 + ksel];
#pragma unroll
      for (int ni = 0; ni < 4; ni++) {
        FragBF fv;
        int dim  = ni * 16 + l16;
        int koff = ks * 32 + half * 8;
        fv.q[0] = *(const uint4*)&Vt[dim][koff];
        fv.q[1] = *(const uint4*)&Vt[dim][koff + 16];
        o[ni] = __builtin_amdgcn_wmma_f32_16x16x32_bf16(
            false, fp.v, false, fv.v, (short)0, o[ni], false, false);
      }
    }
    __syncthreads();  // protect Ks/Vt before next iteration's staging
  }

  // ---- finalize: O /= l, store ctx as bf16 ----
#pragma unroll
  for (int r = 0; r < 8; r++) lrun[r] = 1.0f / lrun[r];
#pragma unroll
  for (int ni = 0; ni < 4; ni++) {
    int col = colh + ni * 16 + l16;
#pragma unroll
    for (int r = 0; r < 8; r++) {
      int row = qrow0 + wave * 16 + r + 8 * half;
      ctx[(size_t)row * D_MODEL + col] = f2bf(o[ni][r] * lrun[r]);
    }
  }
}

// ---------------------------------------------------------------------------
// Launch: convert -> QKV projections -> attention -> output projection
// ---------------------------------------------------------------------------
extern "C" void kernel_launch(void* const* d_in, const int* in_sizes, int n_in,
                              void* d_out, int out_size, void* d_ws,
                              size_t ws_size, hipStream_t stream) {
  const float* x  = (const float*)d_in[0];
  const float* Wq = (const float*)d_in[1];
  const float* bq = (const float*)d_in[2];
  const float* Wk = (const float*)d_in[3];
  const float* bk = (const float*)d_in[4];
  const float* Wv = (const float*)d_in[5];
  const float* bv = (const float*)d_in[6];
  const float* Wo = (const float*)d_in[7];
  const float* bo = (const float*)d_in[8];
  float* out = (float*)d_out;

  char* ws = (char*)d_ws;
  const size_t MB = (size_t)1 << 20;
  unsigned short* xb  = (unsigned short*)(ws + 0 * MB);   // 8 MiB
  unsigned short* Wqb = (unsigned short*)(ws + 8 * MB);   // 2 MiB each
  unsigned short* Wkb = (unsigned short*)(ws + 10 * MB);
  unsigned short* Wvb = (unsigned short*)(ws + 12 * MB);
  unsigned short* Wob = (unsigned short*)(ws + 14 * MB);
  unsigned short* Qb  = (unsigned short*)(ws + 16 * MB);  // 8 MiB each
  unsigned short* Kb  = (unsigned short*)(ws + 24 * MB);
  unsigned short* Vb  = (unsigned short*)(ws + 32 * MB);
  unsigned short* Cb  = (unsigned short*)(ws + 40 * MB);

  const int nx = MTOT * D_MODEL;      // 4,194,304
  const int nw = D_MODEL * D_MODEL;   // 1,048,576
  cvt4<<<nx / 4 / 256, 256, 0, stream>>>((const float4*)x,  (ushort4*)xb,  nx / 4);
  cvt4<<<nw / 4 / 256, 256, 0, stream>>>((const float4*)Wq, (ushort4*)Wqb, nw / 4);
  cvt4<<<nw / 4 / 256, 256, 0, stream>>>((const float4*)Wk, (ushort4*)Wkb, nw / 4);
  cvt4<<<nw / 4 / 256, 256, 0, stream>>>((const float4*)Wv, (ushort4*)Wvb, nw / 4);
  cvt4<<<nw / 4 / 256, 256, 0, stream>>>((const float4*)Wo, (ushort4*)Wob, nw / 4);

  dim3 gg(D_MODEL / 64, MTOT / 128);  // (16, 32)
  const float qscale = 0.125f;        // 1/sqrt(HDIM), folded into Q
  gemm_xwT<true><<<gg, 256, 0, stream>>>(xb, Wqb, bq, Qb, qscale);
  gemm_xwT<true><<<gg, 256, 0, stream>>>(xb, Wkb, bk, Kb, 1.0f);
  gemm_xwT<true><<<gg, 256, 0, stream>>>(xb, Wvb, bv, Vb, 1.0f);

  attn_kernel<<<dim3(SEQ / 128, BATCH * NHEAD), 256, 0, stream>>>(Qb, Kb, Vb, Cb);

  gemm_xwT<false><<<gg, 256, 0, stream>>>(Cb, Wob, bo, out, 1.0f);
}
